// MultiHeadSelfAttention_20873541058921
// MI455X (gfx1250) — compile-verified, matrix operand order checked by reading, not run on hardware
//
#include <hip/hip_runtime.h>
#include <hip/hip_bf16.h>

typedef _Float16 half_t;
typedef __attribute__((ext_vector_type(16))) _Float16 v16h;
typedef __attribute__((ext_vector_type(8)))  _Float16 v8h;
typedef __attribute__((ext_vector_type(8)))  float    v8f;

#define D_MODEL 1024
#define N_HEADS 16
#define D_K     64
#define SEQ     2048
#define BATCH   2
#define M_TOT   (BATCH*SEQ)      /* 4096 */
#define BH      (BATCH*N_HEADS)  /* 32   */

__device__ __forceinline__ v16h cat16(v8h a, v8h b) {
  return __builtin_shufflevector(a, b, 0,1,2,3,4,5,6,7,8,9,10,11,12,13,14,15);
}
__device__ __forceinline__ v8f wmma_f16(v16h a, v16h b, v8f c) {
  return __builtin_amdgcn_wmma_f32_16x16x32_f16(false, a, false, b, (short)0, c,
                                                false, false);
}
__device__ __forceinline__ float fast_exp2(float x) {
#if __has_builtin(__builtin_amdgcn_exp2f)
  return __builtin_amdgcn_exp2f(x);   // raw v_exp_f32; args are in [-inf, 0]
#else
  return exp2f(x);
#endif
}
__device__ __forceinline__ float fast_rcp(float x) {
#if __has_builtin(__builtin_amdgcn_rcpf)
  return __builtin_amdgcn_rcpf(x);
#else
  return 1.0f / x;
#endif
}
// gfx1250 async DMA: global -> LDS, per-lane LDS dest + per-lane 32b offset + SGPR base
__device__ __forceinline__ void async_ld_b128(uint32_t lds_off, uint32_t goff,
                                              const half_t* base) {
  asm volatile("global_load_async_to_lds_b128 %0, %1, %2"
               :: "v"(lds_off), "v"(goff), "s"(base) : "memory");
}
__device__ __forceinline__ void wait_async_le8() {
  asm volatile("s_wait_asynccnt 0x8" ::: "memory");
}
__device__ __forceinline__ void wait_async_0() {
  asm volatile("s_wait_asynccnt 0x0" ::: "memory");
}

// ---------------- conversion kernels ----------------
__global__ __launch_bounds__(256) void cvt_x_kernel(const float* __restrict__ x,
                                                    half_t* __restrict__ xh, int n) {
  int i = blockIdx.x * 256 + threadIdx.x;
  if (i < n) xh[i] = (half_t)x[i];
}

// W[o][i] (f32, 1024x1024) -> Wb[i/32][o][i%32] (f16) : B-matrix blocked layout
__global__ __launch_bounds__(256) void cvt_w_kernel(const float* __restrict__ W,
                                                    half_t* __restrict__ Wb) {
  int idx = blockIdx.x * 256 + threadIdx.x;   // idx = o*1024 + i
  int i = idx & (D_MODEL - 1);
  int o = idx >> 10;
  Wb[(((size_t)(i >> 5) << 10) + o) * 32 + (i & 31)] = (half_t)W[idx];
}

// ------- tiled WMMA GEMM: C[M,N] = A[M,K] * W^T, double-buffered async LDS -------
// 128 threads = 4 waves (2x2), wave tile 64x64, WG tile 128x128, k-step 32
template <typename OUT_T>
__global__ __launch_bounds__(128) void gemm_kernel(const half_t* __restrict__ A,
                                                   const half_t* __restrict__ Bb,
                                                   OUT_T* __restrict__ C,
                                                   int M, int N, int K) {
  __shared__ __align__(16) half_t As[2][128 * 40];   // [m][k] padded stride 40
  __shared__ __align__(16) half_t Bs[2][128 * 40];   // [n][k] padded stride 40
  const int tid  = threadIdx.x;
  const int lane = tid & 31;
  const int wave = tid >> 5;
  const int wr   = wave >> 1;   // 0..1
  const int wc   = wave & 1;    // 0..1
  const int lh   = lane & 15;
  const int lq   = lane >> 4;
  const int m0   = blockIdx.x * 128;
  const int n0   = blockIdx.y * 128;
  const uint32_t LBUF = 128 * 40 * 2;   // LDS bytes per buffer

  // per-thread async copy slots: 4 A-chunks + 4 B-chunks of 8 halfs
  uint32_t aLds[4], bLds[4], aGof[4], bGof[4];
#pragma unroll
  for (int i = 0; i < 4; i++) {
    int id  = tid + i * 128;      // 0..511
    int row = id >> 2;            // 0..127
    int ch  = (id & 3) * 8;
    aLds[i] = (uint32_t)(size_t)&As[0][row * 40 + ch];
    bLds[i] = (uint32_t)(size_t)&Bs[0][row * 40 + ch];
    aGof[i] = (uint32_t)(((m0 + row) * K + ch) * 2);
    bGof[i] = (uint32_t)(((n0 + row) * 32 + ch) * 2);
  }
  auto issue = [&](int k0, int b) {
    uint32_t ak = (uint32_t)(k0 * 2);
    uint32_t bk = (uint32_t)((k0 >> 5) * N * 64);
    uint32_t lb = (uint32_t)b * LBUF;
#pragma unroll
    for (int i = 0; i < 4; i++) {
      async_ld_b128(aLds[i] + lb, aGof[i] + ak, A);
      async_ld_b128(bLds[i] + lb, bGof[i] + bk, Bb);
    }
  };

  v8f acc[4][4] = {};
  const int nk = K / 32;
  issue(0, 0);
  int buf = 0;
  for (int ik = 0; ik < nk; ik++) {
    if (ik + 1 < nk) { issue((ik + 1) * 32, buf ^ 1); wait_async_le8(); }
    else             { wait_async_0(); }
    __syncthreads();

    v16h af[4];
#pragma unroll
    for (int mt = 0; mt < 4; mt++) {
      int r = wr * 64 + mt * 16 + lh;
      af[mt] = cat16(*(const v8h*)(&As[buf][r * 40 + lq * 8]),
                     *(const v8h*)(&As[buf][r * 40 + 16 + lq * 8]));
    }
#pragma unroll
    for (int nt = 0; nt < 4; nt++) {
      int r = wc * 64 + nt * 16 + lh;
      v16h bf = cat16(*(const v8h*)(&Bs[buf][r * 40 + lq * 8]),
                      *(const v8h*)(&Bs[buf][r * 40 + 16 + lq * 8]));
#pragma unroll
      for (int mt = 0; mt < 4; mt++)
        acc[mt][nt] = wmma_f16(af[mt], bf, acc[mt][nt]);
    }
    __syncthreads();
    buf ^= 1;
  }

#pragma unroll
  for (int mt = 0; mt < 4; mt++)
#pragma unroll
    for (int nt = 0; nt < 4; nt++)
#pragma unroll
      for (int r = 0; r < 8; r++) {
        int row = m0 + wr * 64 + mt * 16 + r + lq * 8;
        int col = n0 + wc * 64 + nt * 16 + lh;
        C[(size_t)row * N + col] = (OUT_T)acc[mt][nt][r];
      }
}

// ---------------- RoPE + head reshape + V transpose ----------------
__global__ __launch_bounds__(256) void rope_kernel(
    const half_t* __restrict__ Qh0, const half_t* __restrict__ Kh0,
    const half_t* __restrict__ Vh0, const int* __restrict__ pos,
    half_t* __restrict__ Qh, half_t* __restrict__ Kh, half_t* __restrict__ Vt) {
  int idx = blockIdx.x * 256 + threadIdx.x;   // BH*32*SEQ = 2M
  int s   = idx & (SEQ - 1);
  int k   = (idx >> 11) & 31;
  int bh  = idx >> 16;
  int b = bh >> 4, h = bh & 15;

  size_t src = ((size_t)b * SEQ + s) * D_MODEL + h * D_K + 2 * k;
  float q1 = (float)Qh0[src], q2 = (float)Qh0[src + 1];
  float k1 = (float)Kh0[src], k2 = (float)Kh0[src + 1];
  float v1 = (float)Vh0[src], v2 = (float)Vh0[src + 1];

  float p   = (float)pos[s];
  float inv = __powf(10000.0f, -(float)k * (1.0f / 32.0f));
  float sn, cs;
  __sincosf(p * inv, &sn, &cs);

  // fold (1/sqrt(dk)) * log2(e) into Q so softmax can use exp2 directly
  const float QS = 0.125f * 1.44269504088896f;
  size_t dst = ((size_t)bh * SEQ + s) * D_K + 2 * k;
  Qh[dst]     = (half_t)(QS * (q1 * cs - q2 * sn));
  Qh[dst + 1] = (half_t)(QS * (q1 * sn + q2 * cs));
  Kh[dst]     = (half_t)(k1 * cs - k2 * sn);
  Kh[dst + 1] = (half_t)(k1 * sn + k2 * cs);

  size_t vd = ((size_t)bh * D_K + 2 * k) * SEQ + s;       // V transposed [d][s]
  Vt[vd]       = (half_t)v1;
  Vt[vd + SEQ] = (half_t)v2;
}

// ---------------- causal flash attention (async double-buffered K/V) ----------------
__global__ __launch_bounds__(128) void attn_kernel(
    const half_t* __restrict__ Qh, const half_t* __restrict__ Kh,
    const half_t* __restrict__ Vt, half_t* __restrict__ AOh) {
  __shared__ __align__(16) half_t Ks[2][64 * 72];     // [key][d]  padded 72
  __shared__ __align__(16) half_t Vs[2][64 * 72];     // [d][key]  padded 72
  __shared__ __align__(16) half_t Ps[4][16 * 72];     // per-wave P, [q][key]

  const int bh  = blockIdx.y;
  const int qb  = blockIdx.x * 64;
  const int tid = threadIdx.x;
  const int lane = tid & 31;
  const int wave = tid >> 5;
  const int lh = lane & 15;
  const int lq = lane >> 4;
  const int b = bh >> 4, h = bh & 15;

  const half_t* Qb = Qh + (size_t)bh * SEQ * D_K;
  const half_t* Kb = Kh + (size_t)bh * SEQ * D_K;
  const half_t* Vb = Vt + (size_t)bh * D_K * SEQ;
  const uint32_t KBUF = 64 * 72 * 2;

  const int m0 = qb + wave * 16;

  v16h qf[2];
#pragma unroll
  for (int kk = 0; kk < 2; kk++) {
    const half_t* p = Qb + (size_t)(m0 + lh) * D_K + kk * 32 + lq * 8;
    qf[kk] = cat16(*(const v8h*)p, *(const v8h*)(p + 16));
  }
  v16h ones;
#pragma unroll
  for (int i = 0; i < 16; i++) ones[i] = (half_t)1.0f;

  // per-thread async copy slots (4 K + 4 V chunks of 8 halfs)
  uint32_t kLds[4], vLds[4], kGof[4], vGof[4];
#pragma unroll
  for (int i = 0; i < 4; i++) {
    int id  = tid + i * 128;   // 0..511
    int row = id >> 3;         // 0..63
    int ch  = (id & 7) * 8;
    kLds[i] = (uint32_t)(size_t)&Ks[0][row * 72 + ch];
    vLds[i] = (uint32_t)(size_t)&Vs[0][row * 72 + ch];
    kGof[i] = (uint32_t)((row * D_K + ch) * 2);
    vGof[i] = (uint32_t)((row * SEQ + ch) * 2);
  }
  auto issue = [&](int kc, int bsel) {
    uint32_t lb = (uint32_t)bsel * KBUF;
    uint32_t ko = (uint32_t)(kc * D_K * 2);
    uint32_t vo = (uint32_t)(kc * 2);
#pragma unroll
    for (int i = 0; i < 4; i++) {
      async_ld_b128(kLds[i] + lb, kGof[i] + ko, Kb);
      async_ld_b128(vLds[i] + lb, vGof[i] + vo, Vb);
    }
  };

  const float NEG_INF = -__builtin_inff();
  float mrow[8], lrow[8];
#pragma unroll
  for (int r = 0; r < 8; r++) { mrow[r] = NEG_INF; lrow[r] = 0.f; }
  v8f ox[4] = {};

  const int nchunks = qb / 64 + 1;
  issue(0, 0);
  int buf = 0;
  for (int ci = 0; ci < nchunks; ci++) {
    const int kc = ci * 64;
    if (ci + 1 < nchunks) { issue(kc + 64, buf ^ 1); wait_async_le8(); }
    else                  { wait_async_0(); }
    __syncthreads();

    // scores: 16(q) x 64(key) as 4 C tiles
    v8f sc[4];
#pragma unroll
    for (int nt = 0; nt < 4; nt++) {
      v8f c = {};
#pragma unroll
      for (int kk = 0; kk < 2; kk++) {
        int rr = nt * 16 + lh;
        v16h bfr = cat16(*(const v8h*)(&Ks[buf][rr * 72 + kk * 32 + lq * 8]),
                         *(const v8h*)(&Ks[buf][rr * 72 + kk * 32 + 16 + lq * 8]));
        c = wmma_f16(qf[kk], bfr, c);
      }
      sc[nt] = c;
    }

    // online softmax update; mask only on diagonal-overlapping chunks (wave-uniform)
    const bool needMask = (kc + 63 > m0);
#pragma unroll
    for (int r = 0; r < 8; r++) {
      float mx = mrow[r];
      if (needMask) {
        int q = m0 + r + lq * 8;
#pragma unroll
        for (int nt = 0; nt < 4; nt++) {
          int kcol = kc + nt * 16 + lh;
          float v = sc[nt][r];
          v = (kcol <= q) ? v : NEG_INF;
          sc[nt][r] = v;
          mx = fmaxf(mx, v);
        }
      } else {
#pragma unroll
        for (int nt = 0; nt < 4; nt++) mx = fmaxf(mx, sc[nt][r]);
      }
#pragma unroll
      for (int off = 1; off < 16; off <<= 1)
        mx = fmaxf(mx, __shfl_xor(mx, off, 16));
      float scale = fast_exp2(mrow[r] - mx);
      mrow[r] = mx;
      lrow[r] *= scale;
#pragma unroll
      for (int nt = 0; nt < 4; nt++) {
        float pe = fast_exp2(sc[nt][r] - mx);
        Ps[wave][(r + lq * 8) * 72 + nt * 16 + lh] = (half_t)pe;
      }
#pragma unroll
      for (int dt = 0; dt < 4; dt++) ox[dt][r] *= scale;
    }

    // P (via per-wave LDS, C-layout -> A-layout) times V; row-sums via P x ones
    v8f rs = {};
#pragma unroll
    for (int kk = 0; kk < 2; kk++) {
      const half_t* pp = &Ps[wave][lh * 72 + kk * 32 + lq * 8];
      v16h pf = cat16(*(const v8h*)pp, *(const v8h*)(pp + 16));
      rs = wmma_f16(pf, ones, rs);   // row sums land in lrow's register layout
#pragma unroll
      for (int dt = 0; dt < 4; dt++) {
        int rr = dt * 16 + lh;
        v16h bfr = cat16(*(const v8h*)(&Vs[buf][rr * 72 + kk * 32 + lq * 8]),
                         *(const v8h*)(&Vs[buf][rr * 72 + kk * 32 + 16 + lq * 8]));
        ox[dt] = wmma_f16(pf, bfr, ox[dt]);
      }
    }
#pragma unroll
    for (int r = 0; r < 8; r++) lrow[r] += rs[r];

    __syncthreads();
    buf ^= 1;
  }

  // normalize and write AO as [b][s][h][64] (A-layout for output projection)
#pragma unroll
  for (int r = 0; r < 8; r++) {
    float invl = fast_rcp(lrow[r]);
    int q = m0 + r + lq * 8;
#pragma unroll
    for (int dt = 0; dt < 4; dt++)
      AOh[(((size_t)b * SEQ + q) * N_HEADS + h) * D_K + dt * 16 + lh] =
          (half_t)(ox[dt][r] * invl);
  }
}

// ---------------- launcher ----------------
extern "C" void kernel_launch(void* const* d_in, const int* in_sizes, int n_in,
                              void* d_out, int out_size, void* d_ws, size_t ws_size,
                              hipStream_t stream) {
  (void)in_sizes; (void)n_in; (void)out_size; (void)ws_size;
  const float* x   = (const float*)d_in[0];
  const int*   pos = (const int*)d_in[1];
  const float* WQ  = (const float*)d_in[2];
  const float* WK  = (const float*)d_in[3];
  const float* WV  = (const float*)d_in[4];
  const float* WO  = (const float*)d_in[5];
  float* out = (float*)d_out;

  char* ws = (char*)d_ws;
  size_t off = 0;
  auto alloc = [&](size_t bytes) -> char* {
    char* p = ws + off;
    off += (bytes + 255) & ~(size_t)255;
    return p;
  };
  half_t* xh  = (half_t*)alloc((size_t)M_TOT * D_MODEL * 2);
  half_t* Wqb = (half_t*)alloc((size_t)D_MODEL * D_MODEL * 2);
  half_t* Wkb = (half_t*)alloc((size_t)D_MODEL * D_MODEL * 2);
  half_t* Wvb = (half_t*)alloc((size_t)D_MODEL * D_MODEL * 2);
  half_t* Wob = (half_t*)alloc((size_t)D_MODEL * D_MODEL * 2);
  half_t* Qh0 = (half_t*)alloc((size_t)M_TOT * D_MODEL * 2);
  half_t* Kh0 = (half_t*)alloc((size_t)M_TOT * D_MODEL * 2);
  half_t* Vh0 = (half_t*)alloc((size_t)M_TOT * D_MODEL * 2);
  half_t* Qh  = (half_t*)alloc((size_t)BH * SEQ * D_K * 2);
  half_t* Kh  = (half_t*)alloc((size_t)BH * SEQ * D_K * 2);
  half_t* Vt  = (half_t*)alloc((size_t)BH * SEQ * D_K * 2);
  half_t* AOh = (half_t*)alloc((size_t)M_TOT * D_MODEL * 2);

  cvt_x_kernel<<<(M_TOT * D_MODEL) / 256, 256, 0, stream>>>(x, xh, M_TOT * D_MODEL);
  cvt_w_kernel<<<(D_MODEL * D_MODEL) / 256, 256, 0, stream>>>(WQ, Wqb);
  cvt_w_kernel<<<(D_MODEL * D_MODEL) / 256, 256, 0, stream>>>(WK, Wkb);
  cvt_w_kernel<<<(D_MODEL * D_MODEL) / 256, 256, 0, stream>>>(WV, Wvb);
  cvt_w_kernel<<<(D_MODEL * D_MODEL) / 256, 256, 0, stream>>>(WO, Wob);

  dim3 gg(M_TOT / 128, D_MODEL / 128);
  gemm_kernel<half_t><<<gg, 128, 0, stream>>>(xh, Wqb, Qh0, M_TOT, D_MODEL, D_MODEL);
  gemm_kernel<half_t><<<gg, 128, 0, stream>>>(xh, Wkb, Kh0, M_TOT, D_MODEL, D_MODEL);
  gemm_kernel<half_t><<<gg, 128, 0, stream>>>(xh, Wvb, Vh0, M_TOT, D_MODEL, D_MODEL);

  rope_kernel<<<(BH * 32 * SEQ) / 256, 256, 0, stream>>>(Qh0, Kh0, Vh0, pos, Qh, Kh, Vt);

  attn_kernel<<<dim3(SEQ / 64, BH), 128, 0, stream>>>(Qh, Kh, Vt, AOh);

  gemm_kernel<float><<<gg, 128, 0, stream>>>(AOh, Wob, out, M_TOT, D_MODEL, D_MODEL);
}
